// QGATMultiHeadLayer_88905823027433
// MI455X (gfx1250) — matrix operand order repaired and checked by reading, not compiled, not run on hardware
//
#include <hip/hip_runtime.h>
#include <hip/hip_bf16.h>

typedef __attribute__((ext_vector_type(16))) _Float16 v16h;
typedef __attribute__((ext_vector_type(8)))  _Float16 v8h;
typedef __attribute__((ext_vector_type(8)))  float    v8f;

#define NN     4096
#define IN_F   256
#define OUT_F  64
#define HEADS  4
#define ALPHA  0.2f
#define NEG_INF (-9.0e15f)

// LDS row strides (halves): 280*2=560 B rows -> 16 B aligned, and the 16
// row-lanes hit banks mr*140 words = mr*12 mod 64 (all distinct).
#define XS_STRIDE 280
#define HS_STRIDE 280

// ---------------------------------------------------------------------------
// Kernel A: hamiltonT[h][64][256] (f16, feature-major) from W[h][64][64].
// Transposed so GEMM B-fragments are contiguous per lane.
// ---------------------------------------------------------------------------
__global__ void build_ham_kernel(const float* __restrict__ W, _Float16* __restrict__ hamT) {
    int idx = blockIdx.x * 256 + threadIdx.x;            // HEADS*256*64 = 65536
    int h   = idx >> 14;
    int rem = idx & 16383;
    int r = rem >> 6, c = rem & 63;                      // hamilton row r (K), col c (N)
    int br = r >> 6, rin = r & 63, bc = c >> 4, cin = c & 15;
    const int   comp[4][4] = {{0,1,2,3},{1,0,3,2},{2,3,0,1},{3,2,1,0}};
    const float sgn [4][4] = {{1.f,-1.f,-1.f,-1.f},
                              {1.f, 1.f,-1.f, 1.f},
                              {1.f, 1.f, 1.f,-1.f},
                              {1.f,-1.f, 1.f, 1.f}};
    float v = sgn[br][bc] * W[h*4096 + rin*64 + comp[br][bc]*16 + cin];
    hamT[h*16384 + c*256 + r] = (_Float16)v;             // transposed store
}

// ---------------------------------------------------------------------------
// Kernel B: h = x @ hamilton via v_wmma_f32_16x16x32_f16.
// Stores h row-major (for e GEMVs) AND hT feature-major (for attn B-frags).
// grid (256 row-tiles, HEADS), block 128 = 4 waves (wave = 16-col N-tile)
// ---------------------------------------------------------------------------
__global__ void gemm_h_kernel(const float* __restrict__ x,
                              const _Float16* __restrict__ hamT,
                              _Float16* __restrict__ hout,
                              _Float16* __restrict__ hToutg) {
    __shared__ _Float16 xs [16 * XS_STRIDE];             // x tile, row-major (f16)
    __shared__ _Float16 hsT[64 * HS_STRIDE];             // hamiltonT (feature-major)

    int tile = blockIdx.x, head = blockIdx.y;
    for (int i = threadIdx.x; i < 16 * 256; i += 128) {
        int r = i >> 8, c = i & 255;
        xs[r * XS_STRIDE + c] = (_Float16)x[(tile * 16 + r) * 256 + c];
    }
    const _Float16* hb = hamT + head * 16384;
    for (int i = threadIdx.x; i < 64 * 256; i += 128) {
        int c = i >> 8, r = i & 255;
        hsT[c * HS_STRIDE + r] = hb[i];
    }
    __syncthreads();

    int lane = threadIdx.x & 31;
    int wv   = threadIdx.x >> 5;           // N-tile id
    int lh   = lane >> 4;                  // lane half
    int mr   = lane & 15;
    int k0   = lh * 8;

    v8f acc = {};
#pragma unroll
    for (int k = 0; k < 8; ++k) {
        int kb = k * 32;
        // A 16x32 f16: lane holds row mr, K runs {kb+k0..+7},{kb+16+k0..+7} -> 2x b128
        v8h alo = *(const v8h*)&xs[mr * XS_STRIDE + kb + k0];
        v8h ahi = *(const v8h*)&xs[mr * XS_STRIDE + kb + 16 + k0];
        // B 32x16 f16: lane holds col mr, K runs {kb+8*lh..+7},{kb+16+8*lh..+7} -> 2x b128
        const _Float16* bp = &hsT[(wv * 16 + mr) * HS_STRIDE + kb + 8 * lh];
        v8h blo = *(const v8h*)bp;
        v8h bhi = *(const v8h*)(bp + 16);
        v16h a, b;
#pragma unroll
        for (int t = 0; t < 8; ++t) {
            a[t] = alo[t]; a[t + 8] = ahi[t];
            b[t] = blo[t]; b[t + 8] = bhi[t];
        }
        acc = __builtin_amdgcn_wmma_f32_16x16x32_f16(false, a, false, b,
                                                     (short)0, acc, false, false);
    }
    // C layout: VGPR v -> row v + 8*lh, col = mr
    _Float16* ho  = hout   + (size_t)head * NN * OUT_F + (size_t)(tile * 16) * OUT_F + wv * 16;
    _Float16* hTo = hToutg + (size_t)head * OUT_F * NN + (size_t)(wv * 16 + mr) * NN + tile * 16;
#pragma unroll
    for (int v = 0; v < 8; ++v) {
        _Float16 hv = (_Float16)acc[v];
        ho [(v + 8 * lh) * OUT_F + mr] = hv;
        hTo[ v + 8 * lh            ]   = hv;
    }
}

// ---------------------------------------------------------------------------
// Kernel C1: e_src[h][n], e_dst[h][n]  (tiny GEMVs over row-major h)
// ---------------------------------------------------------------------------
__global__ void calc_e_kernel(const _Float16* __restrict__ hbuf,
                              const float* __restrict__ a_heads,
                              float* __restrict__ e_src, float* __restrict__ e_dst) {
    int tid = blockIdx.x * 128 + threadIdx.x;            // HEADS*NN = 16384
    int h = tid >> 12, n = tid & 4095;
    const _Float16* hv = hbuf + (size_t)h * NN * OUT_F + (size_t)n * OUT_F;
    float s = 0.f, d = 0.f;
#pragma unroll 8
    for (int c = 0; c < OUT_F; ++c) {
        float v = (float)hv[c];
        s += v * a_heads[h * 128 + c];
        d += v * a_heads[h * 128 + 64 + c];
    }
    e_src[h * NN + n] = s;
    e_dst[h * NN + n] = d;
}

// ---------------------------------------------------------------------------
// Kernel C2: per-row softmax stats (max m, 1/sum), all heads fused (adj read
// once). Per-LANE online max/sum (no shuffles in the hot loop); masked
// entries (99.5%) skip both exps. One (m,l) logsumexp-merge tree at the end.
// ---------------------------------------------------------------------------
__global__ void softmax_stats_kernel(const float* __restrict__ adj,
                                     const float* __restrict__ e_src,
                                     const float* __restrict__ e_dst,
                                     float* __restrict__ mrow_out,
                                     float* __restrict__ rl_out) {
    int lane = threadIdx.x & 31;
    int row  = blockIdx.x * 4 + (threadIdx.x >> 5);
    float s[HEADS], m[HEADS], p[HEADS];
#pragma unroll
    for (int h = 0; h < HEADS; ++h) {
        s[h] = e_src[h * NN + row];
        m[h] = -3.0e38f;
        p[h] = 0.f;
    }
    const float* ar = adj + (size_t)row * NN;
    for (int j = lane; j < NN; j += 32) {
        float av = ar[j];
        if (av > 0.f) {                                  // ~0.5% live
#pragma unroll
            for (int h = 0; h < HEADS; ++h) {
                float e = s[h] + e_dst[h * NN + j];
                e = e > 0.f ? e : ALPHA * e;
                float mn = fmaxf(m[h], e);
                p[h] = p[h] * __expf(m[h] - mn) + __expf(e - mn);
                m[h] = mn;
            }
        }
    }
#pragma unroll
    for (int h = 0; h < HEADS; ++h) {
        // merge per-lane (m, p) pairs: logsumexp combine
#pragma unroll
        for (int o = 16; o > 0; o >>= 1) {
            float om = __shfl_xor(m[h], o, 32);
            float op = __shfl_xor(p[h], o, 32);
            float mn = fmaxf(m[h], om);
            p[h] = p[h] * __expf(m[h] - mn) + op * __expf(om - mn);
            m[h] = mn;
        }
        if (lane == 0) {
            mrow_out[h * NN + row] = m[h];
            rl_out[h * NN + row]   = 1.f / p[h];
        }
    }
}

// ---------------------------------------------------------------------------
// Kernel D: fused masked-softmax x h GEMM + ELU.
// grid 256 (16-row tiles), block 128 = 4 waves = 4 heads sharing adj tile.
// P built lane-wise in WMMA A-layout; B-fragments are contiguous b128 loads
// from hT; 1/l applied once at the end via bpermute gather.
// ---------------------------------------------------------------------------
__global__ void attn_out_kernel(const float* __restrict__ adj,
                                const _Float16* __restrict__ hT,
                                const float* __restrict__ e_src,
                                const float* __restrict__ e_dst,
                                const float* __restrict__ mrow,
                                const float* __restrict__ rl,
                                float* __restrict__ out) {
    int tile = blockIdx.x;
    int head = threadIdx.x >> 5;
    int lane = threadIdx.x & 31;
    int lh   = lane >> 4;
    int mr   = lane & 15;
    int rowg = tile * 16 + mr;

    float s  = e_src[head * NN + rowg];
    float mx = mrow [head * NN + rowg];
    float rr = rl   [head * NN + rowg];

    const float*    adjrow = adj + (size_t)rowg * NN;
    const float*    dptr   = e_dst + head * NN;
    const _Float16* hTb    = hT + (size_t)head * OUT_F * NN;

    v8f acc[4] = {{}, {}, {}, {}};
    int k0 = lh * 8;

#pragma unroll 2
    for (int jt = 0; jt < NN / 32; ++jt) {
        int jb = jt * 32;
        __builtin_prefetch(adjrow + jb + 1024 + k0, 0, 0);   // stream adj ahead
        // adj/e_dst: two contiguous 8-float runs per lane (A-layout K sets)
        float4 a0 = *(const float4*)(adjrow + jb + k0);
        float4 a1 = *(const float4*)(adjrow + jb + k0 + 4);
        float4 a2 = *(const float4*)(adjrow + jb + 16 + k0);
        float4 a3 = *(const float4*)(adjrow + jb + 16 + k0 + 4);
        float4 d0 = *(const float4*)(dptr + jb + k0);
        float4 d1 = *(const float4*)(dptr + jb + k0 + 4);
        float4 d2 = *(const float4*)(dptr + jb + 16 + k0);
        float4 d3 = *(const float4*)(dptr + jb + 16 + k0 + 4);
        float av[16] = {a0.x,a0.y,a0.z,a0.w, a1.x,a1.y,a1.z,a1.w,
                        a2.x,a2.y,a2.z,a2.w, a3.x,a3.y,a3.z,a3.w};
        float dv[16] = {d0.x,d0.y,d0.z,d0.w, d1.x,d1.y,d1.z,d1.w,
                        d2.x,d2.y,d2.z,d2.w, d3.x,d3.y,d3.z,d3.w};

        v16h pa;
#pragma unroll
        for (int t = 0; t < 16; ++t) {
            float e = s + dv[t];
            e = e > 0.f ? e : ALPHA * e;
            float pv = av[t] > 0.f ? __expf(e - mx) : 0.f;   // unnormalized att
            pa[t] = (_Float16)pv;
        }
#pragma unroll
        for (int nt = 0; nt < 4; ++nt) {
            // B 32x16: lane holds feat-col (nt*16+mr), K = {jb+8*lh..+7},{jb+16+8*lh..+7}
            const _Float16* bp = hTb + (size_t)(nt * 16 + mr) * NN + jb + 8 * lh;
            v8h blo = *(const v8h*)bp;
            v8h bhi = *(const v8h*)(bp + 16);
            v16h b;
#pragma unroll
            for (int t = 0; t < 8; ++t) { b[t] = blo[t]; b[t + 8] = bhi[t]; }
            acc[nt] = __builtin_amdgcn_wmma_f32_16x16x32_f16(false, pa, false, b,
                                                             (short)0, acc[nt],
                                                             false, false);
        }
    }

    // normalize (1/l per C-row via bpermute gather), ELU, store
#pragma unroll
    for (int v = 0; v < 8; ++v) {
        float f = __shfl(rr, v + 8 * lh, 32);   // rl of row v+8*lh
        int orow = tile * 16 + v + 8 * lh;
#pragma unroll
        for (int nt = 0; nt < 4; ++nt) {
            float val = acc[nt][v] * f;
            val = val > 0.f ? val : (__expf(val) - 1.f);     // ELU
            out[(size_t)orow * (HEADS * OUT_F) + head * OUT_F + nt * 16 + mr] = val;
        }
    }
}

// ---------------------------------------------------------------------------
extern "C" void kernel_launch(void* const* d_in, const int* in_sizes, int n_in,
                              void* d_out, int out_size, void* d_ws, size_t ws_size,
                              hipStream_t stream) {
    const float* x       = (const float*)d_in[0];   // [4096, 256]
    const float* adj     = (const float*)d_in[1];   // [4096, 4096]
    const float* W_heads = (const float*)d_in[2];   // [4, 64, 64]
    const float* a_heads = (const float*)d_in[3];   // [4, 128]
    float* out = (float*)d_out;                     // [4096, 256]

    char* ws = (char*)d_ws;
    _Float16* hamT  = (_Float16*)ws;                          ws += (size_t)HEADS * 256 * 64 * 2;
    _Float16* hbuf  = (_Float16*)ws;                          ws += (size_t)HEADS * NN * OUT_F * 2;
    _Float16* hTbuf = (_Float16*)ws;                          ws += (size_t)HEADS * NN * OUT_F * 2;
    float* e_src = (float*)ws;                                ws += (size_t)HEADS * NN * 4;
    float* e_dst = (float*)ws;                                ws += (size_t)HEADS * NN * 4;
    float* mbuf  = (float*)ws;                                ws += (size_t)HEADS * NN * 4;
    float* rlbuf = (float*)ws;

    build_ham_kernel<<<256, 256, 0, stream>>>(W_heads, hamT);
    gemm_h_kernel<<<dim3(NN / 16, HEADS), 128, 0, stream>>>(x, hamT, hbuf, hTbuf);
    calc_e_kernel<<<(HEADS * NN) / 128, 128, 0, stream>>>(hbuf, a_heads, e_src, e_dst);
    softmax_stats_kernel<<<NN / 4, 128, 0, stream>>>(adj, e_src, e_dst, mbuf, rlbuf);
    attn_out_kernel<<<NN / 16, 128, 0, stream>>>(adj, hTbuf, e_src, e_dst, mbuf, rlbuf, out);
}
